// PointSIS_Encoder_3874060501270
// MI455X (gfx1250) — compile-verified, matrix operand order checked by reading, not run on hardware
//
#include <hip/hip_runtime.h>
#include <hip/hip_bf16.h>
#include <math.h>

// ---------------- problem constants ----------------
#define D_MODEL 256
#define D_INNER 512
#define D_STATE 16
#define DT_RANK 16
#define DEPTH   2
#define N_ORD   2
#define GROUP   1024
#define LEVELS  3
#define BS      2
#define SEQ     (LEVELS * GROUP)   // 3072
#define ROWS    (BS * SEQ)         // 6144 sequence rows
#define NPTS    (BS * GROUP)       // 2048 points per level

typedef unsigned short u16;
typedef unsigned int   u32;
typedef __attribute__((ext_vector_type(16))) __bf16 bf16x16;
typedef __attribute__((ext_vector_type(8)))  float  f32x8;
typedef __attribute__((ext_vector_type(4)))  u32    u32x4;

__device__ __forceinline__ u16 f2bf(float f) {
    u32 u = __float_as_uint(f);
    u32 r = u + 0x7FFFu + ((u >> 16) & 1u);   // round-to-nearest-even
    return (u16)(r >> 16);
}

// ---------------- f32 -> bf16 convert ----------------
__global__ void cvt_bf16_k(const float* __restrict__ src, u16* __restrict__ dst, int n) {
    int i = blockIdx.x * blockDim.x + threadIdx.x;
    if (i < n) dst[i] = f2bf(src[i]);
}

// ---------------- permutation scatter: level feats -> sequence ----------------
__global__ void scatter_in_k(const float* __restrict__ src, const int* __restrict__ order,
                             float* __restrict__ h, int lv) {
    int i = blockIdx.x;        // 0..NPTS-1
    int c = threadIdx.x;       // 0..255
    int o = order[i];
    int b = o >> 10, g = o & (GROUP - 1);
    h[((size_t)(b * SEQ + lv * GROUP + g)) * D_MODEL + c] = src[(size_t)i * D_MODEL + c];
}

// ---------------- residual update ----------------
__global__ void resadd_k(float* __restrict__ resid, const float* __restrict__ h, int n, int add) {
    int i = blockIdx.x * blockDim.x + threadIdx.x;
    if (i < n) resid[i] = add ? (resid[i] + h[i]) : h[i];
}

// ---------------- layernorm (one wave32 per 256-wide row) ----------------
__launch_bounds__(256)
__global__ void ln_k(const float* __restrict__ in, const float* __restrict__ in2,
                     const float* __restrict__ w, const float* __restrict__ bia,
                     u16* __restrict__ out_bf, float* __restrict__ out_f, int rows) {
    int wave = (int)((blockIdx.x * blockDim.x + threadIdx.x) >> 5);
    int lane = threadIdx.x & 31;
    if (wave >= rows) return;
    const float* x  = in + (size_t)wave * D_MODEL;
    const float* x2 = in2 ? (in2 + (size_t)wave * D_MODEL) : nullptr;
    float v[8], s = 0.f, s2 = 0.f;
#pragma unroll
    for (int k = 0; k < 8; ++k) {
        float t = x[lane + 32 * k];
        if (x2) t += x2[lane + 32 * k];
        v[k] = t; s += t; s2 += t * t;
    }
#pragma unroll
    for (int m = 16; m >= 1; m >>= 1) {
        s  += __shfl_xor(s,  m, 32);
        s2 += __shfl_xor(s2, m, 32);
    }
    float mu  = s  * (1.f / D_MODEL);
    float var = s2 * (1.f / D_MODEL) - mu * mu;
    float inv = rsqrtf(var + 1e-5f);
#pragma unroll
    for (int k = 0; k < 8; ++k) {
        int c = lane + 32 * k;
        float o = (v[k] - mu) * inv * w[c] + bia[c];
        if (out_bf) out_bf[(size_t)wave * D_MODEL + c] = f2bf(o);
        if (out_f)  out_f [(size_t)wave * D_MODEL + c] = o;
    }
}

// ---------------- WMMA bf16 GEMM:  C[M,N] = A[M,K] @ W[N,K]^T ----------------
// K compile-time; 2x2 tile blocking (32x32 per wave); fragments via b128 loads.
// A fragment per ISA 7.12.2: j<4 VGPRs = bf16 [hl*8, hl*8+8) ; j>=4 = [16+hl*8, +8).
__device__ __forceinline__ void store_tile(const f32x8& acc, float* __restrict__ C,
                                           u16* __restrict__ Cbf, int ldc,
                                           int m0, int n, bool nvalid, int hl,
                                           float bval, int epilogue) {
    if (!nvalid) return;
#pragma unroll
    for (int r = 0; r < 8; ++r) {
        int m_g = m0 + r + hl * 8;
        float v = acc[r];
        if (epilogue == 1) {                     // softplus(bias + v)
            v += bval;
            v = (v > 20.f) ? v : log1pf(__expf(v));
        }
        size_t off = (size_t)m_g * ldc + n;
        C[off] = v;
        if (Cbf) Cbf[off] = f2bf(v);
    }
}

template <int K>
__launch_bounds__(256)
__global__ void gemm_bf16_k(const u16* __restrict__ A, int lda,
                            const u16* __restrict__ W, int ldw,
                            float* __restrict__ C, u16* __restrict__ Cbf, int ldc,
                            int M, int N,
                            const float* __restrict__ bias, int epilogue) {
    int wave = (int)((blockIdx.x * blockDim.x + threadIdx.x) >> 5);
    int lane = threadIdx.x & 31;
    int bTN = (N + 31) >> 5;                // 32-wide block cols
    int bTM = M >> 5;                       // M is a multiple of 32
    if (wave >= bTM * bTN) return;
    int tm = (wave / bTN) << 5;
    int tn = (wave % bTN) << 5;
    int hl    = lane >> 4;
    int idx16 = lane & 15;

    const u16* Ar0 = A + (size_t)(tm + idx16) * lda;
    const u16* Ar1 = A + (size_t)(tm + 16 + idx16) * lda;
    int n0 = tn + idx16;
    int n1 = tn + 16 + idx16;
    int n0c = n0 < N ? n0 : N - 1;          // clamp: branch-free in-bounds loads
    int n1c = n1 < N ? n1 : N - 1;
    const u16* Wr0 = W + (size_t)n0c * ldw;
    const u16* Wr1 = W + (size_t)n1c * ldw;

    union Frag { bf16x16 v; u32x4 q[2]; };
    const u32x4 zero4 = {0u, 0u, 0u, 0u};
    f32x8 z8 = {0.f, 0.f, 0.f, 0.f, 0.f, 0.f, 0.f, 0.f};
    f32x8 acc00 = z8, acc01 = z8, acc10 = z8, acc11 = z8;

#pragma unroll 4
    for (int kk = 0; kk < K; kk += 32) {
        Frag a0, a1, b0, b1;
        int o0 = kk + hl * 8;               // byte offset 2*o0: 16B aligned
        a0.q[0] = *(const u32x4*)(Ar0 + o0);
        a1.q[0] = *(const u32x4*)(Ar1 + o0);
        b0.q[0] = *(const u32x4*)(Wr0 + o0);
        b1.q[0] = *(const u32x4*)(Wr1 + o0);
        if constexpr (K >= 32) {
            int o1 = o0 + 16;
            a0.q[1] = *(const u32x4*)(Ar0 + o1);
            a1.q[1] = *(const u32x4*)(Ar1 + o1);
            b0.q[1] = *(const u32x4*)(Wr0 + o1);
            b1.q[1] = *(const u32x4*)(Wr1 + o1);
        } else {                             // K==16: upper K-half is zero
            a0.q[1] = zero4; a1.q[1] = zero4;
            b0.q[1] = zero4; b1.q[1] = zero4;
        }
        acc00 = __builtin_amdgcn_wmma_f32_16x16x32_bf16(false, a0.v, false, b0.v,
                                                        (short)0, acc00, false, false);
        acc01 = __builtin_amdgcn_wmma_f32_16x16x32_bf16(false, a0.v, false, b1.v,
                                                        (short)0, acc01, false, false);
        acc10 = __builtin_amdgcn_wmma_f32_16x16x32_bf16(false, a1.v, false, b0.v,
                                                        (short)0, acc10, false, false);
        acc11 = __builtin_amdgcn_wmma_f32_16x16x32_bf16(false, a1.v, false, b1.v,
                                                        (short)0, acc11, false, false);
    }

    float bv0 = bias ? bias[n0c] : 0.f;
    float bv1 = bias ? bias[n1c] : 0.f;
    store_tile(acc00, C, Cbf, ldc, tm,      n0, n0 < N, hl, bv0, epilogue);
    store_tile(acc01, C, Cbf, ldc, tm,      n1, n1 < N, hl, bv1, epilogue);
    store_tile(acc10, C, Cbf, ldc, tm + 16, n0, n0 < N, hl, bv0, epilogue);
    store_tile(acc11, C, Cbf, ldc, tm + 16, n1, n1 < N, hl, bv1, epilogue);
}

// ---------------- causal depthwise conv(4) + SiLU ----------------
__global__ void conv_silu_k(const float* __restrict__ xz,
                            const float* __restrict__ cw, const float* __restrict__ cb,
                            float* __restrict__ xf, u16* __restrict__ xbf) {
    int idx = blockIdx.x * blockDim.x + threadIdx.x;   // ROWS*D_INNER
    if (idx >= ROWS * D_INNER) return;
    int d = idx & (D_INNER - 1);
    int row = idx >> 9;
    int b = row / SEQ, l = row % SEQ;
    float acc = cb[d];
#pragma unroll
    for (int j = 0; j < 4; ++j) {
        int lp = l - 3 + j;
        if (lp >= 0) acc += cw[d * 4 + j] * xz[((size_t)(b * SEQ + lp)) * (2 * D_INNER) + d];
    }
    float o = acc / (1.f + __expf(-acc));   // SiLU
    xf[idx]  = o;
    xbf[idx] = f2bf(o);
}

// ---------------- selective scan (one thread per (batch, channel)) ----------------
__launch_bounds__(256)
__global__ void scan_k(const float* __restrict__ dbl,   // (ROWS,48): [16:32)=B [32:48)=C
                       const float* __restrict__ dt,    // (ROWS,512) softplus'ed
                       const float* __restrict__ x,     // (ROWS,512)
                       const float* __restrict__ Alog,  // (512,16)
                       float* __restrict__ y) {         // (ROWS,512)
    __shared__ float sBC[32 * 32];
    int b = blockIdx.x >> 1;
    int d = ((blockIdx.x & 1) << 8) + threadIdx.x;
    float a[D_STATE], h[D_STATE];
#pragma unroll
    for (int s = 0; s < D_STATE; ++s) { a[s] = -__expf(Alog[d * D_STATE + s]); h[s] = 0.f; }
    for (int tc = 0; tc < SEQ; tc += 32) {
        __syncthreads();
#pragma unroll
        for (int k = 0; k < 4; ++k) {
            int e = threadIdx.x + 256 * k;      // 1024 floats: 32 steps x (B16|C16)
            int t = e >> 5, c = e & 31;
            sBC[e] = dbl[((size_t)(b * SEQ + tc + t)) * 48 + 16 + c];
        }
        __syncthreads();
        for (int t = 0; t < 32; ++t) {
            size_t row = (size_t)(b * SEQ + tc + t);
            float dtv = dt[row * D_INNER + d];
            float xv  = x [row * D_INNER + d];
            float dx  = dtv * xv;
            float yv  = 0.f;
            const float* BC = &sBC[t * 32];
#pragma unroll
            for (int s = 0; s < D_STATE; ++s) {
                float dA = __expf(dtv * a[s]);
                h[s] = dA * h[s] + dx * BC[s];
                yv  += h[s] * BC[16 + s];
            }
            y[row * D_INNER + d] = yv;
        }
    }
}

// ---------------- y = (y + x*Dp) * silu(z) -> bf16 ----------------
__global__ void gate_k(const float* __restrict__ y, const float* __restrict__ xf,
                       const float* __restrict__ Dp, const float* __restrict__ xz,
                       u16* __restrict__ out) {
    int idx = blockIdx.x * blockDim.x + threadIdx.x;
    if (idx >= ROWS * D_INNER) return;
    int d = idx & (D_INNER - 1);
    int row = idx >> 9;
    float v = y[idx] + xf[idx] * Dp[d];
    float z = xz[(size_t)row * (2 * D_INNER) + D_INNER + d];
    out[idx] = f2bf(v * (z / (1.f + __expf(-z))));
}

// ---------------- inverse permutation scatter ----------------
__global__ void scatter_out_k(const float* __restrict__ hfin, const int* __restrict__ inv,
                              float* __restrict__ dst, int lv) {
    int j = blockIdx.x;        // 0..NPTS-1
    int c = threadIdx.x;
    int b = j >> 10, g = j & (GROUP - 1);
    dst[(size_t)inv[j] * D_MODEL + c] =
        hfin[((size_t)(b * SEQ + lv * GROUP + g)) * D_MODEL + c];
}

// ======================= host orchestration =======================
static inline int gemm_blocks(int M, int N) {
    int tiles = (M >> 5) * ((N + 31) >> 5);
    return (tiles + 7) / 8;                 // 8 waves per 256-thread block
}

extern "C" void kernel_launch(void* const* d_in, const int* in_sizes, int n_in,
                              void* d_out, int out_size, void* d_ws, size_t ws_size,
                              hipStream_t stream) {
    const float* feat0     = (const float*)d_in[0];
    const float* feat1     = (const float*)d_in[1];
    const float* feat2     = (const float*)d_in[2];
    const float* norm_w    = (const float*)d_in[3];
    const float* norm_b    = (const float*)d_in[4];
    const float* in_proj_W = (const float*)d_in[5];
    const float* conv_W    = (const float*)d_in[6];
    const float* conv_b    = (const float*)d_in[7];
    const float* x_proj_W  = (const float*)d_in[8];
    const float* dt_proj_W = (const float*)d_in[9];
    const float* dt_proj_b = (const float*)d_in[10];
    const float* A_log     = (const float*)d_in[11];
    const float* D_param   = (const float*)d_in[12];
    const float* out_proj_W= (const float*)d_in[13];
    const float* normf_w   = (const float*)d_in[14];
    const float* normf_b   = (const float*)d_in[15];
    const int*   orders    = (const int*)d_in[16];
    const int*   inverses  = (const int*)d_in[17];
    float* out = (float*)d_out;

    // ---- workspace carve (all 256B aligned) ----
    char* p = (char*)d_ws;
    auto carve = [&](size_t bytes) -> char* {
        char* r = p; p += (bytes + 255) & ~(size_t)255; return r;
    };
    float* lev[3];
    for (int i = 0; i < 3; ++i) lev[i] = (float*)carve((size_t)NPTS * D_MODEL * 4);
    float* h      = (float*)carve((size_t)ROWS * D_MODEL * 4);
    float* resid  = (float*)carve((size_t)ROWS * D_MODEL * 4);
    u16*   hn_bf  = (u16*)  carve((size_t)ROWS * D_MODEL * 2);
    float* xz     = (float*)carve((size_t)ROWS * 2 * D_INNER * 4);  // reused as hfin
    float* xf     = (float*)carve((size_t)ROWS * D_INNER * 4);
    u16*   xbf    = (u16*)  carve((size_t)ROWS * D_INNER * 2);
    float* dbl    = (float*)carve((size_t)ROWS * 48 * 4);
    u16*   dbl_bf = (u16*)  carve((size_t)ROWS * 48 * 2);
    float* dtb    = (float*)carve((size_t)ROWS * D_INNER * 4);
    float* ybuf   = (float*)carve((size_t)ROWS * D_INNER * 4);
    u16*   ygbf   = (u16*)  carve((size_t)ROWS * D_INNER * 2);
    u16*   inW_bf = (u16*)  carve((size_t)(2 * D_INNER) * D_MODEL * 2);
    u16*   xpW_bf = (u16*)  carve((size_t)48 * D_INNER * 2);
    u16*   dtW_bf = (u16*)  carve((size_t)D_INNER * DT_RANK * 2);
    u16*   outW_bf= (u16*)  carve((size_t)D_MODEL * D_INNER * 2);

    auto cvt = [&](const float* s, u16* d, int n) {
        cvt_bf16_k<<<(n + 255) / 256, 256, 0, stream>>>(s, d, n);
    };

    const float* lev_src[3] = {feat2, feat1, feat0};   // reference level_feat order

    for (int o = 0; o < N_ORD; ++o) {
        const int* ord = orders   + o * NPTS;
        const int* inv = inverses + o * NPTS;

        for (int lv = 0; lv < LEVELS; ++lv)
            scatter_in_k<<<NPTS, D_MODEL, 0, stream>>>(
                o == 0 ? lev_src[lv] : lev[lv], ord, h, lv);

        for (int i = 0; i < DEPTH; ++i) {
            int li = o * DEPTH + i;
            resadd_k<<<(ROWS * D_MODEL + 255) / 256, 256, 0, stream>>>(
                resid, h, ROWS * D_MODEL, i);
            ln_k<<<ROWS / 8, 256, 0, stream>>>(resid, nullptr,
                norm_w + (size_t)li * D_MODEL, norm_b + (size_t)li * D_MODEL,
                hn_bf, nullptr, ROWS);
            cvt(in_proj_W  + (size_t)li * 2 * D_INNER * D_MODEL, inW_bf, 2 * D_INNER * D_MODEL);
            cvt(x_proj_W   + (size_t)li * 48 * D_INNER,          xpW_bf, 48 * D_INNER);
            cvt(dt_proj_W  + (size_t)li * D_INNER * DT_RANK,     dtW_bf, D_INNER * DT_RANK);
            cvt(out_proj_W + (size_t)li * D_MODEL * D_INNER,     outW_bf, D_MODEL * D_INNER);
            // xz = hn @ inW^T    (6144x256 @ 256x1024), K=256
            gemm_bf16_k<D_MODEL><<<gemm_blocks(ROWS, 2 * D_INNER), 256, 0, stream>>>(
                hn_bf, D_MODEL, inW_bf, D_MODEL, xz, nullptr, 2 * D_INNER,
                ROWS, 2 * D_INNER, nullptr, 0);
            conv_silu_k<<<(ROWS * D_INNER + 255) / 256, 256, 0, stream>>>(
                xz, conv_W + (size_t)li * D_INNER * 4, conv_b + (size_t)li * D_INNER,
                xf, xbf);
            // dbl = x @ xpW^T    (6144x512 @ 512x48), K=512, bf16 shadow
            gemm_bf16_k<D_INNER><<<gemm_blocks(ROWS, 48), 256, 0, stream>>>(
                xbf, D_INNER, xpW_bf, D_INNER, dbl, dbl_bf, 48,
                ROWS, 48, nullptr, 0);
            // dt = softplus(dt_raw @ dtW^T + dtb), K=16 (upper half zero-filled)
            gemm_bf16_k<DT_RANK><<<gemm_blocks(ROWS, D_INNER), 256, 0, stream>>>(
                dbl_bf, 48, dtW_bf, DT_RANK, dtb, nullptr, D_INNER,
                ROWS, D_INNER, dt_proj_b + (size_t)li * D_INNER, 1);
            scan_k<<<4, 256, 0, stream>>>(dbl, dtb, xf,
                A_log + (size_t)li * D_INNER * D_STATE, ybuf);
            gate_k<<<(ROWS * D_INNER + 255) / 256, 256, 0, stream>>>(
                ybuf, xf, D_param + (size_t)li * D_INNER, xz, ygbf);
            // h = ygated @ outW^T  (6144x512 @ 512x256), K=512
            gemm_bf16_k<D_INNER><<<gemm_blocks(ROWS, D_MODEL), 256, 0, stream>>>(
                ygbf, D_INNER, outW_bf, D_INNER, h, nullptr, D_MODEL,
                ROWS, D_MODEL, nullptr, 0);
        }
        ln_k<<<ROWS / 8, 256, 0, stream>>>(h, resid,
            normf_w + (size_t)o * D_MODEL, normf_b + (size_t)o * D_MODEL,
            nullptr, xz, ROWS);
        for (int lv = 0; lv < LEVELS; ++lv) {
            float* dst = (o == N_ORD - 1) ? (out + (size_t)lv * NPTS * D_MODEL) : lev[lv];
            scatter_out_k<<<NPTS, D_MODEL, 0, stream>>>(xz, inv, dst, lv);
        }
    }
}
